// MultiHeadSelfAttention_32452772889293
// MI455X (gfx1250) — matrix operand order repaired and checked by reading, not compiled, not run on hardware
//
#include <hip/hip_runtime.h>
#include <hip/hip_bf16.h>

// Problem constants
#define BB 4
#define SS 2048
#define DD 768
#define HH 12
#define DHD 64
#define MTOT (BB * SS)   // 8192

typedef __attribute__((ext_vector_type(16))) __bf16 v16bf;
typedef __attribute__((ext_vector_type(8)))  float  v8f;
typedef __attribute__((ext_vector_type(8)))  __bf16 bf16x8;
typedef __attribute__((ext_vector_type(4)))  __bf16 bf16x4;

typedef unsigned int u32;
typedef __attribute__((ext_vector_type(4))) u32 u32x4;
typedef __attribute__((ext_vector_type(8))) int  i32x8;
typedef __attribute__((ext_vector_type(4))) int  i32x4;

// Toolchain heuristic: therock-10.0 headers (6-arg TDM builtin) ship the TDM
// descriptor header; ROCm 7.2 (5-arg builtin) does not.
#if __has_include(<hip/amd_detail/amd_gfx1250_TDM.h>)
#define TDM_SIX_ARGS 1
#else
#define TDM_SIX_ARGS 0
#endif

__device__ __forceinline__ v16bf cat8(bf16x8 lo, bf16x8 hi) {
    return __builtin_shufflevector(lo, hi, 0, 1, 2, 3, 4, 5, 6, 7,
                                           8, 9, 10, 11, 12, 13, 14, 15);
}

__device__ __forceinline__ v8f wmma_bf16(v16bf a, v16bf b, v8f c) {
    return __builtin_amdgcn_wmma_f32_16x16x32_bf16(
        /*neg_a=*/false, a, /*neg_b=*/false, b,
        /*c_mod=*/(short)0, c, /*reuse_a=*/false, /*reuse_b=*/false);
}

// ---------------------------------------------------------------------------
// Tensor Data Mover: 2D tile load Global -> LDS (D# per CDNA5 ISA ch.8).
// dsize_code: 0=1B, 1=2B, 2=4B elements. Dims/strides in elements.
// ---------------------------------------------------------------------------
__device__ __forceinline__ void tdm_load_2d(u32 lds_byte_off, const void* gptr,
                                            u32 dsize_code,
                                            u32 tensor_d0, u32 tensor_d1,
                                            u32 tile_d0, u32 tile_d1,
                                            u32 stride_d0)
{
    const unsigned long long ga = (unsigned long long)(uintptr_t)gptr;

    u32x4 g0;
    g0[0] = 1u;                                             // count=1 (valid)
    g0[1] = lds_byte_off;                                   // lds_addr
    g0[2] = (u32)(ga & 0xFFFFFFFFull);                      // global_addr[31:0]
    g0[3] = (u32)((ga >> 32) & 0x1FFFFFFull) | (2u << 30);  // addr[56:32]|type=2

    i32x8 g1;
    g1[0] = (int)(dsize_code << 16);                        // data_size
    g1[1] = (int)((tensor_d0 & 0xFFFFu) << 16);             // tensor_dim0 lo
    g1[2] = (int)((tensor_d0 >> 16) | ((tensor_d1 & 0xFFFFu) << 16));
    g1[3] = (int)((tensor_d1 >> 16) | (tile_d0 << 16));     // dim1 hi | tile_dim0
    g1[4] = (int)(tile_d1 & 0xFFFFu);                       // tile_dim1 (dim2=0)
    g1[5] = (int)stride_d0;                                 // dim0_stride[31:0]
    g1[6] = 0;
    g1[7] = 0;

    i32x4 z4 = {0, 0, 0, 0};
#if TDM_SIX_ARGS
    i32x8 z8 = {};
    __builtin_amdgcn_tensor_load_to_lds(g0, g1, z4, z4, z8, 0);
#else
    __builtin_amdgcn_tensor_load_to_lds(g0, g1, z4, z4, 0);
#endif
}

__device__ __forceinline__ u32 lds_offset(const void* p) {
    // Flat LDS-aperture pointers carry the LDS byte offset in the low 32 bits.
    return (u32)(uintptr_t)p;
}

// ---------------------------------------------------------------------------
// Kernel 0a: X f32 -> bf16 (same layout). 4 elements/thread.
// ---------------------------------------------------------------------------
__global__ __launch_bounds__(256) void cvt_x_kernel(const float* __restrict__ src,
                                                    __bf16* __restrict__ dst) {
    const size_t i = ((size_t)blockIdx.x * blockDim.x + threadIdx.x) * 4;
    const float4 f = *(const float4*)(src + i);
    bf16x4 o;
    o.x = (__bf16)f.x; o.y = (__bf16)f.y; o.z = (__bf16)f.z; o.w = (__bf16)f.w;
    *(bf16x4*)(dst + i) = o;
}

// ---------------------------------------------------------------------------
// Kernel 0b: W [K,N] f32 -> Wt [N,K] bf16 (transpose so B operands are
// contiguous 16-element segments in K).
// ---------------------------------------------------------------------------
__global__ __launch_bounds__(256) void cvt_wt_kernel(const float* __restrict__ W,
                                                     __bf16* __restrict__ Wt) {
    const int tid = blockIdx.x * blockDim.x + threadIdx.x;   // = n*DD + k
    const int n = tid / DD;
    const int k = tid % DD;
    Wt[tid] = (__bf16)W[k * DD + n];
}

// ---------------------------------------------------------------------------
// Generic inner GEMM stage: block 256 thr = 8 waves; block tile 128(M)x64(N);
// K-step 64 (two WMMA k-chunks); double-buffered TDM staging.
// sA: [128][64] bf16 row-major; sB: [64 n][64 k] bf16 (from transposed W).
// ---------------------------------------------------------------------------
__device__ __forceinline__ void gemm_stage(const __bf16* __restrict__ sA,
                                           const __bf16* __restrict__ sB,
                                           int wave, int lane, v8f acc[4])
{
    const int mrow = lane & 15;
    const int half = lane >> 4;
#pragma unroll
    for (int c = 0; c < 2; ++c) {
        const __bf16* arow = &sA[(wave * 16 + mrow) * 64 + 32 * c + 16 * half];
        const v16bf a = cat8(*(const bf16x8*)arow, *(const bf16x8*)(arow + 8));
#pragma unroll
        for (int g = 0; g < 4; ++g) {
            const int nl = g * 16 + (lane & 15);
            const v16bf bm = *(const v16bf*)&sB[nl * 64 + 32 * c + 16 * half];
            acc[g] = wmma_bf16(a, bm, acc[g]);
        }
    }
}

// ---------------------------------------------------------------------------
// Kernel 1: fused QKV projection (all-bf16 operands, TDM double-buffered).
// Q,K stored bf16 [B,H,S,DH]; V stored transposed bf16 [B,H,DH,S].
// ---------------------------------------------------------------------------
__global__ __launch_bounds__(256) void qkv_proj_kernel(
    const __bf16* __restrict__ Xbf,
    const __bf16* __restrict__ Wqt, const float* __restrict__ bq,
    const __bf16* __restrict__ Wkt, const float* __restrict__ bk,
    const __bf16* __restrict__ Wvt, const float* __restrict__ bvv,
    __bf16* __restrict__ Qbf, __bf16* __restrict__ Kbf, __bf16* __restrict__ Vt)
{
    __shared__ __align__(32) __bf16 sX[2][128 * 64];  // 2 x 16 KB
    __shared__ __align__(32) __bf16 sW[2][64 * 64];   // 2 x  8 KB

    const int lane = threadIdx.x & 31;
    const int wave = threadIdx.x >> 5;

    const int BLOCKS_PER_P = (MTOT / 128) * (DD / 64);    // 64*12 = 768
    const int p     = blockIdx.x / BLOCKS_PER_P;          // 0=Q 1=K 2=V
    const int rem   = blockIdx.x % BLOCKS_PER_P;
    const int mTile = rem / (DD / 64);
    const int nTile = rem % (DD / 64);

    const __bf16* Wt   = (p == 0) ? Wqt : (p == 1) ? Wkt : Wvt;
    const float*  bias = (p == 0) ? bq  : (p == 1) ? bk  : bvv;

    const __bf16* Asrc = Xbf + (size_t)mTile * 128 * DD;
    const __bf16* Bsrc = Wt  + (size_t)nTile * 64 * DD;

    v8f acc[4];
#pragma unroll
    for (int g = 0; g < 4; ++g) acc[g] = (v8f){};

    if (threadIdx.x < 32) {   // wave 0 drives the TDM
        tdm_load_2d(lds_offset(sX[0]), Asrc, 1, DD, MTOT, 64, 128, DD);
        tdm_load_2d(lds_offset(sW[0]), Bsrc, 1, DD, DD,   64, 64,  DD);
    }

    const int NSTEP = DD / 64;   // 12
    for (int s = 0; s < NSTEP; ++s) {
        const int cur = s & 1;
        if (threadIdx.x < 32) {
            if (s + 1 < NSTEP) {
                tdm_load_2d(lds_offset(sX[cur ^ 1]), Asrc + (s + 1) * 64,
                            1, DD, MTOT, 64, 128, DD);
                tdm_load_2d(lds_offset(sW[cur ^ 1]), Bsrc + (s + 1) * 64,
                            1, DD, DD, 64, 64, DD);
                __builtin_amdgcn_s_wait_tensorcnt(2);   // current buffers ready
            } else {
                __builtin_amdgcn_s_wait_tensorcnt(0);
            }
        }
        __syncthreads();
        gemm_stage(sX[cur], sW[cur], wave, lane, acc);
        __syncthreads();   // all waves done before this buffer is re-filled
    }

    const int half = lane >> 4;
#pragma unroll
    for (int g = 0; g < 4; ++g) {
        const int ncol = nTile * 64 + g * 16 + (lane & 15);
        const float bval = bias[ncol];
        const int hh = ncol >> 6;       // head
        const int dd = ncol & 63;       // depth within head
#pragma unroll
        for (int v = 0; v < 8; ++v) {
            const float cval = acc[g][v] + bval;   // C row = v+8*half
            const int r  = mTile * 128 + wave * 16 + v + 8 * half;
            const int bb = r >> 11;
            const int ss = r & (SS - 1);
            if (p == 2) {
                Vt[(((size_t)bb * HH + hh) * DHD + dd) * SS + ss] = (__bf16)cval;
            } else {
                __bf16* dst = (p == 0) ? Qbf : Kbf;
                dst[(((size_t)bb * HH + hh) * SS + ss) * DHD + dd] = (__bf16)cval;
            }
        }
    }
}

// ---------------------------------------------------------------------------
// Kernel 2: flash attention. One wave = 16 queries of one (b,h); online
// softmax over 32-key blocks; all WMMA operands are wide contiguous loads.
// ---------------------------------------------------------------------------
__global__ __launch_bounds__(128) void attention_kernel(
    const __bf16* __restrict__ Qbf, const __bf16* __restrict__ Kbf,
    const __bf16* __restrict__ Vt,  const float* __restrict__ mask,
    __bf16* __restrict__ Ctx)
{
    __shared__ __align__(32) __bf16 ldsP[4][16 * 32];  // per-wave P relayout

    const int lane  = threadIdx.x & 31;
    const int wave  = threadIdx.x >> 5;
    const int gw    = blockIdx.x * 4 + wave;
    const int bh    = gw >> 7;            // / (S/16)
    const int qTile = gw & 127;
    const int b     = bh / HH;
    const int h     = bh % HH;

    const __bf16* Qh    = Qbf + (size_t)bh * SS * DHD;
    const __bf16* Kh    = Kbf + (size_t)bh * SS * DHD;
    const __bf16* Vh    = Vt  + (size_t)bh * DHD * SS;
    const float*  maskb = mask + (size_t)b * SS;

    const int mrow = lane & 15;
    const int half = lane >> 4;

    // Q tile (16x64) as two A operands (k-chunks of 32): contiguous segments
    const __bf16* qrow = Qh + (size_t)(qTile * 16 + mrow) * DHD + 16 * half;
    const v16bf aQ0 = cat8(*(const bf16x8*)qrow,        *(const bf16x8*)(qrow + 8));
    const v16bf aQ1 = cat8(*(const bf16x8*)(qrow + 32), *(const bf16x8*)(qrow + 40));

    v8f acc[4];
#pragma unroll
    for (int j = 0; j < 4; ++j) acc[j] = (v8f){};
    float mrv[8], lrv[8];
#pragma unroll
    for (int v = 0; v < 8; ++v) { mrv[v] = -1e30f; lrv[v] = 0.0f; }

    for (int kb = 0; kb < SS; kb += 32) {
        // ---- logits S = Q * K^T: B operand = 16 contiguous bf16 of K row ----
        v8f s0 = {}, s1 = {};
#pragma unroll
        for (int t = 0; t < 2; ++t) {
            const int key = kb + 16 * t + (lane & 15);   // B col = key
            const __bf16* krow = Kh + (size_t)key * DHD + 16 * half;
            const v16bf kb0 = *(const v16bf*)krow;        // dh 16h..16h+15
            const v16bf kb1 = *(const v16bf*)(krow + 32); // dh 32+16h..
            if (t == 0) { s0 = wmma_bf16(aQ0, kb0, s0); s0 = wmma_bf16(aQ1, kb1, s0); }
            else        { s1 = wmma_bf16(aQ0, kb0, s1); s1 = wmma_bf16(aQ1, kb1, s1); }
        }

        // ---- scale + mask + online softmax (rows live in lane-halves) ----
        const float mk0 = maskb[kb +      (lane & 15)] * -1e9f;
        const float mk1 = maskb[kb + 16 + (lane & 15)] * -1e9f;
#pragma unroll
        for (int v = 0; v < 8; ++v) {
            float e0 = s0[v] * 0.125f + mk0;
            float e1 = s1[v] * 0.125f + mk1;
            float rm = fmaxf(e0, e1);
#pragma unroll
            for (int off = 8; off >= 1; off >>= 1)
                rm = fmaxf(rm, __shfl_xor(rm, off, 32));
            const float mn    = fmaxf(mrv[v], rm);
            const float alpha = __expf(mrv[v] - mn);
            const float p0 = __expf(e0 - mn);
            const float p1 = __expf(e1 - mn);
            float ps = p0 + p1;
#pragma unroll
            for (int off = 8; off >= 1; off >>= 1)
                ps += __shfl_xor(ps, off, 32);
            lrv[v] = lrv[v] * alpha + ps;
            mrv[v] = mn;
            s0[v] = p0; s1[v] = p1;
#pragma unroll
            for (int j = 0; j < 4; ++j) acc[j][v] *= alpha;
        }

        // ---- P: C-layout -> A-layout via per-wave LDS bounce ----
        __bf16* P = ldsP[wave];
#pragma unroll
        for (int v = 0; v < 8; ++v) {
            const int r = v + 8 * half;
            P[r * 32 +      (lane & 15)] = (__bf16)s0[v];
            P[r * 32 + 16 + (lane & 15)] = (__bf16)s1[v];
        }
        asm volatile("s_wait_dscnt 0" ::: "memory");
        const __bf16* prow = &P[mrow * 32 + 16 * half];
        const v16bf aP = cat8(*(const bf16x8*)prow, *(const bf16x8*)(prow + 8));

        // ---- ctx += P(16x32) * V(32x64): V^T rows are contiguous in keys ----
#pragma unroll
        for (int j = 0; j < 4; ++j) {
            const int dh = 16 * j + (lane & 15);     // B col = dh
            const v16bf bV = *(const v16bf*)(Vh + (size_t)dh * SS + kb + 16 * half);
            acc[j] = wmma_bf16(aP, bV, acc[j]);
        }
    }

    // ---- normalize and write ctx (concat-head layout [B,S,D], bf16) ----
#pragma unroll
    for (int v = 0; v < 8; ++v) {
        const float inv  = 1.0f / lrv[v];
        const int   srow = qTile * 16 + v + 8 * half;
        const size_t base = ((size_t)(b * SS + srow)) * DD + h * DHD;
#pragma unroll
        for (int j = 0; j < 4; ++j) {
            Ctx[base + 16 * j + (lane & 15)] = (__bf16)(acc[j][v] * inv);
        }
    }
}

// ---------------------------------------------------------------------------
// Kernel 3: output projection: Out = Ctx(bf16) x Wo + bo, same tiling as qkv.
// ---------------------------------------------------------------------------
__global__ __launch_bounds__(256) void out_proj_kernel(
    const __bf16* __restrict__ Ctx, const __bf16* __restrict__ Wot,
    const float* __restrict__ bo, float* __restrict__ Out)
{
    __shared__ __align__(32) __bf16 sA[2][128 * 64];
    __shared__ __align__(32) __bf16 sW[2][64 * 64];

    const int lane = threadIdx.x & 31;
    const int wave = threadIdx.x >> 5;

    const int mTile = blockIdx.x / (DD / 64);
    const int nTile = blockIdx.x % (DD / 64);

    const __bf16* Asrc = Ctx + (size_t)mTile * 128 * DD;
    const __bf16* Bsrc = Wot + (size_t)nTile * 64 * DD;

    v8f acc[4];
#pragma unroll
    for (int g = 0; g < 4; ++g) acc[g] = (v8f){};

    if (threadIdx.x < 32) {
        tdm_load_2d(lds_offset(sA[0]), Asrc, 1, DD, MTOT, 64, 128, DD);
        tdm_load_2d(lds_offset(sW[0]), Bsrc, 1, DD, DD,   64, 64,  DD);
    }

    const int NSTEP = DD / 64;
    for (int s = 0; s < NSTEP; ++s) {
        const int cur = s & 1;
        if (threadIdx.x < 32) {
            if (s + 1 < NSTEP) {
                tdm_load_2d(lds_offset(sA[cur ^ 1]), Asrc + (s + 1) * 64,
                            1, DD, MTOT, 64, 128, DD);
                tdm_load_2d(lds_offset(sW[cur ^ 1]), Bsrc + (s + 1) * 64,
                            1, DD, DD, 64, 64, DD);
                __builtin_amdgcn_s_wait_tensorcnt(2);
            } else {
                __builtin_amdgcn_s_wait_tensorcnt(0);
            }
        }
        __syncthreads();
        gemm_stage(sA[cur], sW[cur], wave, lane, acc);
        __syncthreads();
    }

    const int half = lane >> 4;
#pragma unroll
    for (int g = 0; g < 4; ++g) {
        const int ncol = nTile * 64 + g * 16 + (lane & 15);
        const float bval = bo[ncol];
#pragma unroll
        for (int v = 0; v < 8; ++v) {
            const int r = mTile * 128 + wave * 16 + v + 8 * half;
            Out[(size_t)r * DD + ncol] = acc[g][v] + bval;
        }
    }
}

// ---------------------------------------------------------------------------
extern "C" void kernel_launch(void* const* d_in, const int* in_sizes, int n_in,
                              void* d_out, int out_size, void* d_ws, size_t ws_size,
                              hipStream_t stream) {
    (void)in_sizes; (void)n_in; (void)out_size; (void)ws_size;
    const float* X    = (const float*)d_in[0];
    const float* mask = (const float*)d_in[1];
    const float* wq   = (const float*)d_in[2];
    const float* bq   = (const float*)d_in[3];
    const float* wk   = (const float*)d_in[4];
    const float* bk   = (const float*)d_in[5];
    const float* wv   = (const float*)d_in[6];
    const float* bvv  = (const float*)d_in[7];
    const float* wo   = (const float*)d_in[8];
    const float* bo   = (const float*)d_in[9];
    float* out = (float*)d_out;

    const size_t nElem = (size_t)BB * HH * SS * DHD;  // 6,291,456
    const size_t wElem = (size_t)DD * DD;             //   589,824
    __bf16* Qbf = (__bf16*)d_ws;
    __bf16* Kbf = Qbf + nElem;
    __bf16* Vt  = Kbf + nElem;
    __bf16* Ctx = Vt  + nElem;
    __bf16* Xbf = Ctx + nElem;
    __bf16* Wqt = Xbf + nElem;
    __bf16* Wkt = Wqt + wElem;
    __bf16* Wvt = Wkt + wElem;
    __bf16* Wot = Wvt + wElem;                        // total ~68 MB

    // One-time precision/layout conversion
    cvt_x_kernel<<<6144, 256, 0, stream>>>(X, Xbf);           // 6.29M elems /4
    cvt_wt_kernel<<<2304, 256, 0, stream>>>(wq, Wqt);
    cvt_wt_kernel<<<2304, 256, 0, stream>>>(wk, Wkt);
    cvt_wt_kernel<<<2304, 256, 0, stream>>>(wv, Wvt);
    cvt_wt_kernel<<<2304, 256, 0, stream>>>(wo, Wot);

    // 3 projections x (64 x 12) 128x64 blocks
    qkv_proj_kernel<<<2304, 256, 0, stream>>>(Xbf, Wqt, bq, Wkt, bk, Wvt, bvv,
                                              Qbf, Kbf, Vt);
    // 48 heads x 128 q-tiles, 4 waves/block
    attention_kernel<<<1536, 128, 0, stream>>>(Qbf, Kbf, Vt, mask, Ctx);
    // 64 x 12 blocks of 128x64
    out_proj_kernel<<<768, 256, 0, stream>>>(Ctx, Wot, bo, out);
}
